// DERM_H_49048526520489
// MI455X (gfx1250) — compile-verified, matrix operand order checked by reading, not run on hardware
//
#include <hip/hip_runtime.h>
#include <hip/hip_bf16.h>
#include <math.h>

// ---------------------------------------------------------------------------
// Problem constants (match reference)
// ---------------------------------------------------------------------------
#define B_  4096
#define L_  50
#define C_  100
#define T_  50000
#define D_  128
#define H1_ 256
#define H2_ 128

typedef __attribute__((ext_vector_type(16))) __bf16 v16bf;
typedef __attribute__((ext_vector_type(8)))  __bf16 v8bf;
typedef __attribute__((ext_vector_type(8)))  float  v8f;

__device__ __forceinline__ unsigned short f32_to_bf16_rne(float f) {
    unsigned u = __float_as_uint(f);
    u += 0x7FFFu + ((u >> 16) & 1u);
    return (unsigned short)(u >> 16);
}

// WMMA 16x16x32 bf16 fragment coordinate maps (per CDNA5 ISA 7.12.2).
// A (16xK tile of 32): lane = 16*half + m ; element i: k = (i>>3)*16 + half*8 + (i&7)
// B fragment uses the mirrored map with n in place of m; we pre-pack weights so
// each lane reads its 16 bf16 values as one contiguous 32-byte load.
__device__ __forceinline__ void bfrag_coords(int lane, int i, int& k, int& n) {
    n = lane & 15;
    const int half  = lane >> 4;
    const int chunk = i >> 3;
    k = chunk * 16 + half * 8 + (i & 7);
}

// ---------------------------------------------------------------------------
// Kernel 1: masked-mean pooling of user/item tag embeddings -> x bf16 [B, 2D]
// ---------------------------------------------------------------------------
__global__ void __launch_bounds__(128) pool_kernel(
    const int* __restrict__ utags, const int* __restrict__ ulen,
    const int* __restrict__ itags, const int* __restrict__ ilen,
    const float* __restrict__ table,
    unsigned short* __restrict__ x /* bf16 [B, 2D] */)
{
    const int b = blockIdx.x;
    const int d = threadIdx.x;            // 0..127 == D channel
    __shared__ int s_idx[2 * L_];
    if (threadIdx.x < L_) {
        s_idx[threadIdx.x]       = utags[b * L_ + threadIdx.x];
        s_idx[L_ + threadIdx.x]  = itags[b * L_ + threadIdx.x];
    }
    __syncthreads();
    const int ul = ulen[b];
    const int il = ilen[b];
    float su = 0.f, si = 0.f;
    #pragma unroll 5
    for (int l = 0; l < L_; ++l) {
        if (l < ul) su += table[(size_t)s_idx[l]      * D_ + d];
        if (l < il) si += table[(size_t)s_idx[L_ + l] * D_ + d];
    }
    su /= (float)(ul > 1 ? ul : 1);
    si /= (float)(il > 1 ? il : 1);
    x[(size_t)b * (2 * D_) + d]       = f32_to_bf16_rne(su);
    x[(size_t)b * (2 * D_) + D_ + d]  = f32_to_bf16_rne(si);
}

// ---------------------------------------------------------------------------
// Kernel 2: repack W1 [256x256] and W2 [256x128] (row-major f32, K x N) into
// bf16 WMMA B-fragments: fragment (kt, nt) = 512 bf16; lane reads 16 contiguous.
// ---------------------------------------------------------------------------
__global__ void __launch_bounds__(256) pack_kernel(
    const float* __restrict__ W1, const float* __restrict__ W2,
    unsigned short* __restrict__ W1p, unsigned short* __restrict__ W2p)
{
    const int idx = blockIdx.x * blockDim.x + threadIdx.x;
    const int NW1 = (2 * D_ / 32) * (H1_ / 16) * 512;   // 8*16*512 = 65536
    const int NW2 = (H1_ / 32) * (H2_ / 16) * 512;      // 8*8*512  = 32768
    if (idx < NW1) {
        const int f = idx >> 9, r = idx & 511;
        const int lane = r >> 4, i = r & 15;
        const int kt = f / (H1_ / 16), nt = f % (H1_ / 16);
        int k, n; bfrag_coords(lane, i, k, n);
        W1p[idx] = f32_to_bf16_rne(W1[(size_t)(kt * 32 + k) * H1_ + nt * 16 + n]);
    } else if (idx < NW1 + NW2) {
        const int j = idx - NW1;
        const int f = j >> 9, r = j & 511;
        const int lane = r >> 4, i = r & 15;
        const int kt = f / (H2_ / 16), nt = f % (H2_ / 16);
        int k, n; bfrag_coords(lane, i, k, n);
        W2p[j] = f32_to_bf16_rne(W2[(size_t)(kt * 32 + k) * H2_ + nt * 16 + n]);
    }
}

// ---------------------------------------------------------------------------
// Kernel 3: tag_score[t] = dot(tag_table[t], Wr[H2 : H2+D])  (rank-1 trick)
// ---------------------------------------------------------------------------
__global__ void __launch_bounds__(128) tagscore_kernel(
    const float* __restrict__ table, const float* __restrict__ Wr,
    float* __restrict__ tscore)
{
    const int t = blockIdx.x;
    const int d = threadIdx.x;     // 0..127
    float p = table[(size_t)t * D_ + d] * Wr[H2_ + d];
    // wave32 reduction
    #pragma unroll
    for (int off = 16; off > 0; off >>= 1) p += __shfl_down(p, off);
    __shared__ float s[4];
    const int w = threadIdx.x >> 5, ln = threadIdx.x & 31;
    if (ln == 0) s[w] = p;
    __syncthreads();
    if (threadIdx.x == 0) tscore[t] = s[0] + s[1] + s[2] + s[3];
}

// ---------------------------------------------------------------------------
// Kernel 4: fused 2-layer MLP on 16-row tiles with bf16 WMMA, collapsing the
// output straight to mid_score[b] = dot(relu(layer2), Wr[0:H2]).
// Block = 256 threads (8 waves), grid = B/16.
// ---------------------------------------------------------------------------
__global__ void __launch_bounds__(256) mlp_kernel(
    const unsigned short* __restrict__ x,    // bf16 [B, 2D]
    const unsigned short* __restrict__ W1p,  // packed bf16 fragments
    const float* __restrict__ b1,
    const unsigned short* __restrict__ W2p,  // packed bf16 fragments
    const float* __restrict__ b2,
    const float* __restrict__ Wr,            // f32 [2D] (first H2 used here)
    float* __restrict__ mid_score)           // f32 [B]
{
    __shared__ unsigned short lds_h[16 * H1_];   // relu(layer1) in bf16, 8 KB
    __shared__ float lds_score[16];

    const int tid  = threadIdx.x;
    const int wv   = tid >> 5;
    const int lane = tid & 31;
    const int half = lane >> 4;
    const int mrow = lane & 15;
    const int b0   = blockIdx.x * 16;

    if (tid < 16) lds_score[tid] = 0.f;

    // ---- layer 1: h = relu(x @ W1 + b1); wave wv owns cols [wv*32, wv*32+32)
    v8f acc0 = {0.f,0.f,0.f,0.f,0.f,0.f,0.f,0.f};
    v8f acc1 = {0.f,0.f,0.f,0.f,0.f,0.f,0.f,0.f};
    const int nt0 = wv * 2;                      // first of two 16-col tiles
    #pragma unroll
    for (int kt = 0; kt < 8; ++kt) {
        const int kb = kt * 32;
        const unsigned short* ap = x + (size_t)(b0 + mrow) * (2 * D_) + kb + half * 8;
        union { v16bf v; v8bf h[2]; } af;
        af.h[0] = *(const v8bf*)(ap);
        af.h[1] = *(const v8bf*)(ap + 16);
        const v16bf bf0 = *(const v16bf*)(W1p + (((size_t)kt * 16 + nt0)     << 9) + lane * 16);
        const v16bf bf1 = *(const v16bf*)(W1p + (((size_t)kt * 16 + nt0 + 1) << 9) + lane * 16);
        acc0 = __builtin_amdgcn_wmma_f32_16x16x32_bf16(false, af.v, false, bf0, (short)0, acc0, false, false);
        acc1 = __builtin_amdgcn_wmma_f32_16x16x32_bf16(false, af.v, false, bf1, (short)0, acc1, false, false);
    }
    {
        const int n_a = nt0 * 16 + (lane & 15);
        const int n_b = n_a + 16;
        const float bb0 = b1[n_a], bb1 = b1[n_b];
        #pragma unroll
        for (int r = 0; r < 8; ++r) {
            const int m = r + half * 8;
            float h0 = acc0[r] + bb0; h0 = h0 > 0.f ? h0 : 0.f;
            float h1 = acc1[r] + bb1; h1 = h1 > 0.f ? h1 : 0.f;
            lds_h[m * H1_ + n_a] = f32_to_bf16_rne(h0);
            lds_h[m * H1_ + n_b] = f32_to_bf16_rne(h1);
        }
    }
    __syncthreads();

    // ---- layer 2: mid = relu(h @ W2 + b2); wave wv owns cols [wv*16, wv*16+16)
    v8f acc = {0.f,0.f,0.f,0.f,0.f,0.f,0.f,0.f};
    #pragma unroll
    for (int kt = 0; kt < 8; ++kt) {
        const int kb = kt * 32;
        const unsigned short* ap = lds_h + mrow * H1_ + kb + half * 8;
        union { v16bf v; v8bf h[2]; } af;
        af.h[0] = *(const v8bf*)(ap);
        af.h[1] = *(const v8bf*)(ap + 16);
        const v16bf bfr = *(const v16bf*)(W2p + (((size_t)kt * 8 + wv) << 9) + lane * 16);
        acc = __builtin_amdgcn_wmma_f32_16x16x32_bf16(false, af.v, false, bfr, (short)0, acc, false, false);
    }
    {
        const int n = wv * 16 + (lane & 15);
        const float bb = b2[n], wr = Wr[n];   // Wr[0:H2] multiplies mid
        #pragma unroll
        for (int r = 0; r < 8; ++r) {
            const int m = r + half * 8;
            float v = acc[r] + bb; v = v > 0.f ? v : 0.f;
            atomicAdd(&lds_score[m], v * wr);   // ds_add_f32
        }
    }
    __syncthreads();
    if (tid < 16) mid_score[b0 + tid] = lds_score[tid];
}

// ---------------------------------------------------------------------------
// Kernel 5: score[b,c] = sigmoid(mid_score[b] + tag_score[candi[b,c]] + br)
// ---------------------------------------------------------------------------
__global__ void __launch_bounds__(256) score_kernel(
    const int* __restrict__ candi, const float* __restrict__ mid_score,
    const float* __restrict__ tscore, const float* __restrict__ br,
    float* __restrict__ out)
{
    const int t = blockIdx.x * blockDim.x + threadIdx.x;
    if (t >= B_ * C_) return;
    const int b = t / C_;
    const float z = mid_score[b] + tscore[candi[t]] + br[0];
    out[t] = 1.f / (1.f + __expf(-z));
}

// ---------------------------------------------------------------------------
// Host-side launcher
// ---------------------------------------------------------------------------
extern "C" void kernel_launch(void* const* d_in, const int* in_sizes, int n_in,
                              void* d_out, int out_size, void* d_ws, size_t ws_size,
                              hipStream_t stream) {
    const int*   user_tags = (const int*)  d_in[0];
    const int*   user_len  = (const int*)  d_in[1];
    const int*   item_tags = (const int*)  d_in[2];
    const int*   item_len  = (const int*)  d_in[3];
    const int*   candi     = (const int*)  d_in[4];
    const float* table     = (const float*)d_in[5];
    const float* W1        = (const float*)d_in[6];
    const float* b1        = (const float*)d_in[7];
    const float* W2        = (const float*)d_in[8];
    const float* b2        = (const float*)d_in[9];
    const float* Wr        = (const float*)d_in[10];
    const float* br        = (const float*)d_in[11];
    float* out = (float*)d_out;

    // Workspace layout (all 256B-aligned, ~2.5 MB total)
    char* ws = (char*)d_ws;
    unsigned short* xb   = (unsigned short*)(ws);                       // 4096*256*2 = 2,097,152
    unsigned short* W1p  = (unsigned short*)(ws + 2097152);             // 131,072
    unsigned short* W2p  = (unsigned short*)(ws + 2097152 + 131072);    // 65,536
    float* tscore        = (float*)(ws + 2097152 + 131072 + 65536);     // 200,000 (pad to 200,704)
    float* mscore        = (float*)(ws + 2097152 + 131072 + 65536 + 200704); // 16,384

    // 1) pooled embeddings -> bf16 x
    pool_kernel<<<B_, 128, 0, stream>>>(user_tags, user_len, item_tags, item_len, table, xb);

    // 2) weight repack into WMMA fragment order
    pack_kernel<<<(65536 + 32768) / 256, 256, 0, stream>>>(W1, W2, W1p, W2p);

    // 3) per-tag scalar scores (rank-1 Wr decomposition)
    tagscore_kernel<<<T_, 128, 0, stream>>>(table, Wr, tscore);

    // 4) fused MLP with bf16 WMMA -> per-row scalar
    mlp_kernel<<<B_ / 16, 256, 0, stream>>>(xb, W1p, b1, W2p, b2, Wr, mscore);

    // 5) final gather + sigmoid
    score_kernel<<<(B_ * C_ + 255) / 256, 256, 0, stream>>>(candi, mscore, tscore, br, out);
}
